// ConvblockWithTarget_14130442404232
// MI455X (gfx1250) — matrix operand order repaired and checked by reading, not compiled, not run on hardware
//
#include <hip/hip_runtime.h>

// ---------------------------------------------------------------------------
// Fused conv-block for MI455X (gfx1250, wave32).
//
// Strategy: the dominant cost is the (B*16372 x 64) @ (64 x 448) gate GEMM
// (7.5 GFLOP f32).  Memory floor is ~50MB / 23.3 TB/s ~= 2.2us, so f32 WMMA
// (16x16x4) would be ~8x too slow on instruction count.  We use a split-
// precision scheme: x = xh + xl, W = Wh + Wl (f16 RNE splits) and compute
// xh*Wh + xl*Wh + xh*Wl with v_wmma_f32_16x16x32_f16 (f32 accum), giving
// ~f32 accuracy at 3x the minimal f16-WMMA instruction count.
//
// Round-1: hardware V_TANH_F32 (or branch-free exp2/rcp fallback) instead of
// the divergent libm tanhf; LayerNorm stats parallelized with wave32
// __shfl_xor butterflies.
// Round-2: stage the x tile with CDNA5 async DMA (GLOBAL_LOAD_ASYNC_TO_LDS_
// B128, ASYNCcnt + s_wait_asynccnt) instead of global_load->ds_store VGPR
// round-trips.
// ---------------------------------------------------------------------------

typedef _Float16 v16h __attribute__((ext_vector_type(16)));
typedef _Float16 v8h  __attribute__((ext_vector_type(8)));
typedef float    v8f  __attribute__((ext_vector_type(8)));

#define B_  8
#define L_  16384
#define C_  64
#define K_  7
#define T_  8186            // output positions per batch = (L - 2K)/2 + 1

#define TT_ 64              // output t positions per block
#define XROWS_ 140          // x rows needed: 2*TT + 2K - 2
#define XS_STR 68           // xs row stride (floats), padded vs 64 banks
#define G_STR  116          // gate buffer row stride (floats)

// LDS layout (bytes), dynamic shared:
//   xs   : 140*68*4  = 38,080   f32 x tile
//   ah   : 128*64*2  = 16,384   f16 hi of gate-input rows (A-matrix rows)
//   al   : 128*64*2  = 16,384   f16 lo
//   gbuf : 128*116*4 = 59,392   f32 tanh(gates) for one 16-channel d-stage
//          (reused after the stage loop as ynf: 64*68*4 = 17,408)
//   ybuf : 64*68*4   = 17,408   f32 pre-LN activations
// total = 147,648  -> 2 blocks per WGP (320KB LDS)
#define OFF_AH   38080
#define OFF_AL   (38080 + 16384)
#define OFF_GBUF (38080 + 32768)
#define OFF_YBUF (OFF_GBUF + 59392)
#define SMEM_BYTES (OFF_YBUF + 17408)

// Hardware tanh: gfx1250 has V_TANH_F32 (transcendental).  Fallback is
// branch-free: tanh(|x|) = (1-e)/(1+e), e = exp2(-2*log2(e)*|x|) -> 0 for
// large |x| (no overflow path, no divergence), then copysign.
__device__ __forceinline__ float fast_tanh(float x) {
#if __has_builtin(__builtin_amdgcn_tanhf)
  return __builtin_amdgcn_tanhf(x);
#else
  const float ax = __builtin_fabsf(x);
  const float e  = __builtin_amdgcn_exp2f(ax * -2.8853900817779268f); // -2*log2(e)
  const float t  = (1.0f - e) * __builtin_amdgcn_rcpf(1.0f + e);
  return __builtin_copysignf(t, x);
#endif
}

// Async global->LDS DMA of 16 bytes for this lane (CDNA5 VGLOBAL opcode 98,
// GV mode: LDS address from the vdst VGPR, memory address from a VGPR pair).  The LDS
// byte offset is the low 32 bits of the shared->generic pointer cast.
// Tracked by ASYNCcnt.
__device__ __forceinline__ void async_copy_b128(void* lds_dst, const void* gsrc) {
  const unsigned ldsOff = (unsigned)(size_t)lds_dst;
  const unsigned long long ga = (unsigned long long)(size_t)gsrc;
  asm volatile("global_load_async_to_lds_b128 %0, %1, off"
               :: "v"(ldsOff), "v"(ga) : "memory");
}

__device__ __forceinline__ void async_drain() {
#if __has_builtin(__builtin_amdgcn_s_wait_asynccnt)
  __builtin_amdgcn_s_wait_asynccnt(0);
#else
  asm volatile("s_wait_asynccnt 0x0" ::: "memory");
#endif
}

// Build a 16x32 f16 A-fragment in this code's contiguous k order (matches packed B):
// lane L holds row m = L&15; element s -> k = (L>>4)*16 + s
// => lane's 16 halves are one contiguous 16-element k-run at k = h*16
// relative to the 32-wide k-chunk.  `rowbase` points at the 64-f16 row start.
__device__ __forceinline__ v16h load_afrag(const _Float16* rowbase, int kc, int h) {
  const _Float16* p = rowbase + kc * 32 + h * 16;
  v8h p0 = *(const v8h*)(p);
  v8h p1 = *(const v8h*)(p + 8);
  v16h r;
  *(v8h*)&r       = p0;
  *((v8h*)&r + 1) = p1;
  return r;
}

// ---------------------------------------------------------------------------
// Prep kernel: pack weights (C,C,K) and conv_kernel (C,C) into B-fragment
// order, f16 hi/lo.  B-matrix (32x16 K x N) per-lane layout (this code's
// order): lane L = column n (L&15), k-run = (L>>4)*16 .. +16 contiguous.
// Packed so each lane's fragment is one contiguous 32-byte load:
//   pos(n, c) = (((nt*2 + kc)*32) + hh*16 + (n&15))*16 + (c&15)
// with nt = n>>4, kc = c>>5, hh = (c>>4)&1.   Gate GEMM uses n = kk*64 + d.
// ---------------------------------------------------------------------------
__global__ void cb_prep_pack(const float* __restrict__ w, const float* __restrict__ ck,
                             _Float16* __restrict__ wh, _Float16* __restrict__ wl,
                             _Float16* __restrict__ ch, _Float16* __restrict__ cl) {
  const int id = blockIdx.x * 256 + threadIdx.x;
  if (id < 448 * 64) {
    const int n = id >> 6, c = id & 63;        // n = kk*64 + d
    const int kk = n >> 6, d = n & 63;
    const float v = w[d * 448 + c * 7 + kk];   // weights[d][c][kk]
    const int nt = n >> 4, nl = n & 15;
    const int kc = c >> 5, hh = (c >> 4) & 1, j = c & 15;
    const int pos = (((nt * 2 + kc) * 32) + hh * 16 + nl) * 16 + j;
    const _Float16 hi = (_Float16)v;
    wh[pos] = hi;
    wl[pos] = (_Float16)(v - (float)hi);
  } else if (id < 448 * 64 + 64 * 64) {
    const int i2 = id - 448 * 64;
    const int c = i2 >> 6, o = i2 & 63;        // z[t][o] = sum_c yn[t][c]*ck[c][o]
    const float v = ck[c * 64 + o];
    const int nt = o >> 4, nl = o & 15;
    const int kc = c >> 5, hh = (c >> 4) & 1, j = c & 15;
    const int pos = (((nt * 2 + kc) * 32) + hh * 16 + nl) * 16 + j;
    const _Float16 hi = (_Float16)v;
    ch[pos] = hi;
    cl[pos] = (_Float16)(v - (float)hi);
  }
}

// ---------------------------------------------------------------------------
// Main fused kernel. grid = (128, 8), block = 256 threads = 8 wave32.
// Block (bx, b): outputs t in [64*bx, 64*bx+64) of batch b.
// ---------------------------------------------------------------------------
__global__ __launch_bounds__(256)
void cb_fused(const float* __restrict__ x,
              const _Float16* __restrict__ wpk_hi, const _Float16* __restrict__ wpk_lo,
              const _Float16* __restrict__ ckpk_hi, const _Float16* __restrict__ ckpk_lo,
              const float* __restrict__ ln_scale, const float* __restrict__ ln_bias,
              const float* __restrict__ conv_bias, const float* __restrict__ prelu_slope,
              float* __restrict__ out) {
  extern __shared__ char smem[];
  float*    xs   = (float*)smem;
  _Float16* ah   = (_Float16*)(smem + OFF_AH);
  _Float16* al   = (_Float16*)(smem + OFF_AL);
  float*    gbuf = (float*)(smem + OFF_GBUF);
  float*    ybuf = (float*)(smem + OFF_YBUF);
  float*    ynf  = gbuf;  // overlay: gbuf is dead after the d-stage loop

  const int tid  = threadIdx.x;
  const int lane = tid & 31;
  const int wave = tid >> 5;
  const int h    = lane >> 4;       // half-wave group
  const int nl   = lane & 15;

  const int b   = blockIdx.y;
  const int t0  = blockIdx.x * TT_;
  const int xr0 = 2 * t0;           // first x row needed (taps); gate rows are xr0+12..xr0+139

  // ---- Phase 1: stage x tile into LDS via async DMA (zero-fill past end)
  {
    const float* xbat = x + (size_t)b * L_ * C_;
    for (int i = tid; i < XROWS_ * 16; i += 256) {
      const int row = i >> 4, c4 = i & 15;
      float* dst = xs + row * XS_STR + c4 * 4;
      const int gr = xr0 + row;
      if (gr < L_) {
        async_copy_b128(dst, xbat + (size_t)gr * C_ + c4 * 4);
      } else {
        *(float4*)dst = make_float4(0.f, 0.f, 0.f, 0.f);
      }
    }
    async_drain();
  }
  __syncthreads();

  // ---- Phase 2: f16 hi/lo split of the 128 gate-input rows (xs rows 12..139)
  for (int i = tid; i < 128 * 64; i += 256) {
    const int m = i >> 6, c = i & 63;
    const float f = xs[(m + 12) * XS_STR + c];
    const _Float16 hi = (_Float16)f;
    ah[m * 64 + c] = hi;
    al[m * 64 + c] = (_Float16)(f - (float)hi);
  }
  __syncthreads();

  // ---- Phase 3: each wave loads A-fragments for its 16 gate rows (kept in VGPRs)
  v16h aH[2], aL[2];
  {
    const _Float16* rbh = ah + (wave * 16 + nl) * 64;
    const _Float16* rbl = al + (wave * 16 + nl) * 64;
#pragma unroll
    for (int kc = 0; kc < 2; ++kc) {
      aH[kc] = load_afrag(rbh, kc, h);
      aL[kc] = load_afrag(rbl, kc, h);
    }
  }

  // ---- Phase 4: 4 d-stages; per stage compute 7 n-tiles (n = kk*64 + d),
  //      tanh, stage to LDS, then gated tap-sum + residual into ybuf.
  for (int dt = 0; dt < 4; ++dt) {
#pragma unroll
    for (int kk = 0; kk < 7; ++kk) {
      const int nt = kk * 4 + dt;
      v8f acc = {};
#pragma unroll
      for (int kc = 0; kc < 2; ++kc) {
        const int boff = ((nt * 2 + kc) * 32 + lane) * 16;
        const v16h bH = *(const v16h*)(wpk_hi + boff);
        const v16h bL = *(const v16h*)(wpk_lo + boff);
        acc = __builtin_amdgcn_wmma_f32_16x16x32_f16(false, aH[kc], false, bH, (short)0, acc, false, false);
        acc = __builtin_amdgcn_wmma_f32_16x16x32_f16(false, aL[kc], false, bH, (short)0, acc, false, false);
        acc = __builtin_amdgcn_wmma_f32_16x16x32_f16(false, aH[kc], false, bL, (short)0, acc, false, false);
      }
      // C/D frag: lane L, vgpr v -> m = v + (L>>4)*8, n-local = L&15
      const int gr = wave * 16 + h * 8;
      const int gc = kk * 16 + nl;
#pragma unroll
      for (int v = 0; v < 8; ++v) gbuf[(gr + v) * G_STR + gc] = fast_tanh(acc[v]);
    }
    __syncthreads();

    // gated tap-sum for channels d = dt*16 .. dt*16+15
    for (int i = tid; i < TT_ * 16; i += 256) {
      const int dloc = i & 15, tt = i >> 4;
      const int d = dt * 16 + dloc;
      const int lr = 2 * tt;
      float y = xs[(lr + 12) * XS_STR + d];  // residual x[b, 12+2t, d]
#pragma unroll
      for (int kk = 0; kk < 7; ++kk) {
#pragma unroll
        for (int s = 0; s < 2; ++s) {
          y += xs[(lr + 2 * kk + s) * XS_STR + d] * gbuf[(lr + s) * G_STR + kk * 16 + dloc];
        }
      }
      ybuf[tt * XS_STR + d] = y;
    }
    __syncthreads();
  }

  // ---- Phase 5: LayerNorm over C=64, fully parallel: 4 lanes per row each
  //      reduce 16 channels; combine with two wave32 shuffle butterflies
  //      (the 4 partials sit in adjacent lanes of the same wave).
  {
    const int r = tid >> 2;          // row 0..63
    const int q = tid & 3;           // quarter 0..3
    float s0 = 0.f, s1 = 0.f;
#pragma unroll
    for (int j = 0; j < 16; ++j) {
      const float v = ybuf[r * XS_STR + q * 16 + j];
      s0 += v;
      s1 += v * v;
    }
    s0 += __shfl_xor(s0, 1); s0 += __shfl_xor(s0, 2);
    s1 += __shfl_xor(s1, 1); s1 += __shfl_xor(s1, 2);
    const float mu  = s0 * (1.0f / 64.0f);
    const float var = s1 * (1.0f / 64.0f) - mu * mu;
    const float inv = rsqrtf(var + 1e-6f);
#pragma unroll
    for (int j = 0; j < 16; ++j) {
      const int c = q * 16 + j;
      const float yn = (ybuf[r * XS_STR + c] - mu) * inv * ln_scale[c] + ln_bias[c];
      ynf[r * XS_STR + c] = yn;
      const _Float16 hi = (_Float16)yn;
      ah[r * 64 + c] = hi;
      al[r * 64 + c] = (_Float16)(yn - (float)hi);
    }
  }
  __syncthreads();

  // ---- Phase 6: pointwise conv (64x64) via split-f16 WMMA + bias + PReLU +
  //      residual (out = yn + z), guarded coalesced stores.
  const float slope = prelu_slope[0];
#pragma unroll
  for (int rep = 0; rep < 2; ++rep) {
    const int p  = wave * 2 + rep;     // 16 (mt,nt) tile pairs over 8 waves
    const int mt = p >> 2, nt = p & 3;
    const _Float16* rbh = ah + (mt * 16 + nl) * 64;
    const _Float16* rbl = al + (mt * 16 + nl) * 64;
    v8f acc = {};
#pragma unroll
    for (int kc = 0; kc < 2; ++kc) {
      const v16h a2H = load_afrag(rbh, kc, h);
      const v16h a2L = load_afrag(rbl, kc, h);
      const int boff = ((nt * 2 + kc) * 32 + lane) * 16;
      const v16h bH = *(const v16h*)(ckpk_hi + boff);
      const v16h bL = *(const v16h*)(ckpk_lo + boff);
      acc = __builtin_amdgcn_wmma_f32_16x16x32_f16(false, a2H, false, bH, (short)0, acc, false, false);
      acc = __builtin_amdgcn_wmma_f32_16x16x32_f16(false, a2L, false, bH, (short)0, acc, false, false);
      acc = __builtin_amdgcn_wmma_f32_16x16x32_f16(false, a2H, false, bL, (short)0, acc, false, false);
    }
    const int n = nt * 16 + nl;
    const float bias = conv_bias[n];
    const int rbase = mt * 16 + h * 8;
#pragma unroll
    for (int v = 0; v < 8; ++v) {
      const int row = rbase + v;
      float z = acc[v] + bias;
      z = (z >= 0.f) ? z : slope * z;
      const float o = ynf[row * XS_STR + n] + z;
      const int t = t0 + row;
      if (t < T_) out[((size_t)b * T_ + t) * C_ + n] = o;
    }
  }
}

// ---------------------------------------------------------------------------
extern "C" void kernel_launch(void* const* d_in, const int* in_sizes, int n_in,
                              void* d_out, int out_size, void* d_ws, size_t ws_size,
                              hipStream_t stream) {
  (void)in_sizes; (void)n_in; (void)out_size; (void)ws_size;
  const float* x     = (const float*)d_in[0];
  const float* w     = (const float*)d_in[1];
  const float* ln_s  = (const float*)d_in[2];
  const float* ln_b  = (const float*)d_in[3];
  const float* ck    = (const float*)d_in[4];
  const float* cb    = (const float*)d_in[5];
  const float* slope = (const float*)d_in[6];

  // workspace: packed f16 hi/lo weight fragments (131,072 bytes total)
  _Float16* wpk_hi  = (_Float16*)d_ws;
  _Float16* wpk_lo  = wpk_hi + 448 * 64;
  _Float16* ckpk_hi = wpk_lo + 448 * 64;
  _Float16* ckpk_lo = ckpk_hi + 64 * 64;

  cb_prep_pack<<<(448 * 64 + 64 * 64 + 255) / 256, 256, 0, stream>>>(
      w, ck, wpk_hi, wpk_lo, ckpk_hi, ckpk_lo);

  dim3 grid((T_ + TT_ - 1) / TT_, B_);  // (128, 8)
  cb_fused<<<grid, 256, SMEM_BYTES, stream>>>(
      x, wpk_hi, wpk_lo, ckpk_hi, ckpk_lo, ln_s, ln_b, cb, slope, (float*)d_out);
}
